// Qwen3Attention_65120294142552
// MI455X (gfx1250) — matrix operand order repaired and checked by reading, not compiled, not run on hardware
//
#include <hip/hip_runtime.h>

#define BATCH 2
#define SEQ   2048
#define HID   2048
#define NH    16
#define NKV   8
#define HD    128

typedef __attribute__((ext_vector_type(16))) __bf16 v16bf;
typedef __attribute__((ext_vector_type(8)))  __bf16 v8bf;
typedef __attribute__((ext_vector_type(8)))  float  v8f;
typedef __attribute__((ext_vector_type(4)))  float  v4f;

static __device__ __forceinline__ v8f wmma_bf16(v16bf a, v16bf b, v8f c) {
  return __builtin_amdgcn_wmma_f32_16x16x32_bf16(false, a, false, b, (short)0, c,
                                                 false, false);
}

// A fragment (16x32, M x K): p = &A[row][kchunk + (lane>>4)*8]; loads K {0..7, 16..23}
static __device__ __forceinline__ v16bf frag_a_bf(const __bf16* p) {
  v8bf lo = *(const v8bf*)p;
  v8bf hi = *(const v8bf*)(p + 16);
  return __builtin_shufflevector(lo, hi, 0, 1, 2, 3, 4, 5, 6, 7,
                                 8, 9, 10, 11, 12, 13, 14, 15);
}
// B fragment (32x16, K x N) from row-major (N x K): p = &W[col][kchunk + (lane>>4)*8]; loads K {0..7, 16..23}
static __device__ __forceinline__ v16bf frag_b_bf(const __bf16* p) {
  v8bf lo = *(const v8bf*)p;
  v8bf hi = *(const v8bf*)(p + 16);
  return __builtin_shufflevector(lo, hi, 0, 1, 2, 3, 4, 5, 6, 7,
                                 8, 9, 10, 11, 12, 13, 14, 15);
}

// ---------------------------------------------------------------------------
// Elementwise fp32 -> bf16 (one-time precision change; 8 elems / lane)
// ---------------------------------------------------------------------------
__global__ __launch_bounds__(256) void cvt_f32_bf16(const float* __restrict__ X,
                                                    __bf16* __restrict__ Y) {
  const size_t i = ((size_t)blockIdx.x * 256 + threadIdx.x) * 8;
  v4f x0 = *(const v4f*)(X + i);
  v4f x1 = *(const v4f*)(X + i + 4);
  v8bf y;
#pragma unroll
  for (int j = 0; j < 4; ++j) { y[j] = (__bf16)x0[j]; y[4 + j] = (__bf16)x1[j]; }
  *(v8bf*)(Y + i) = y;
}

// ---------------------------------------------------------------------------
// C(M,N) = A(M,K) * W(N,K)^T, bf16 in / fp32 out.
// 32x64 tile per wave (2 A-frags x 4 B-frags -> 8 WMMA per K-chunk of 32).
// Block = 8 waves covering 64 rows x 256 cols. grid = (N/256, M/64).
// ---------------------------------------------------------------------------
__global__ __launch_bounds__(256) void gemm_nt_bf16(
    const __bf16* __restrict__ A, const __bf16* __restrict__ W,
    float* __restrict__ C, int M, int N, int K) {
  const int lane  = threadIdx.x & 31;
  const int wave  = threadIdx.x >> 5;
  const int mbase = blockIdx.y * 64 + (wave >> 2) * 32;
  const int nbase = blockIdx.x * 256 + (wave & 3) * 64;
  const int ka = (lane >> 4) * 8;
  const int kb = (lane >> 4) * 8;

  const __bf16* a0 = A + (size_t)(mbase + (lane & 15)) * K + ka;
  const __bf16* a1 = a0 + (size_t)16 * K;
  const __bf16* w0 = W + (size_t)(nbase + (lane & 15)) * K + kb;

  v8f acc[2][4] = {};
  for (int k = 0; k < K; k += 32) {
    if (k + 256 < K) {
      __builtin_prefetch(a0 + k + 256, 0, 1);
      __builtin_prefetch(w0 + k + 256, 0, 1);
    }
    v16bf af0 = frag_a_bf(a0 + k);
    v16bf af1 = frag_a_bf(a1 + k);
#pragma unroll
    for (int j = 0; j < 4; ++j) {
      v16bf bf = frag_b_bf(w0 + (size_t)(j * 16) * K + k);
      acc[0][j] = wmma_bf16(af0, bf, acc[0][j]);
      acc[1][j] = wmma_bf16(af1, bf, acc[1][j]);
    }
  }
#pragma unroll
  for (int i = 0; i < 2; ++i)
#pragma unroll
    for (int r = 0; r < 8; ++r) {
      const int crow = mbase + 16 * i + r + 8 * (lane >> 4);
      float* cp = C + (size_t)crow * N + nbase + (lane & 15);
#pragma unroll
      for (int j = 0; j < 4; ++j) cp[j * 16] = acc[i][j][r];
    }
}

// ---------------------------------------------------------------------------
// RMSNorm over HD + RoPE; one wave per (b,s,head). Writes bf16 (B,NHx,S,HD).
// ---------------------------------------------------------------------------
__global__ __launch_bounds__(256) void normrope(
    const float* __restrict__ X,     // [B*S, NHx*HD]
    const float* __restrict__ nw,    // [HD]
    const float* __restrict__ cosb,  // [B*S, HD]
    const float* __restrict__ sinb,  // [B*S, HD]
    __bf16* __restrict__ Y,          // [B, NHx, S, HD]
    int NHx, float outscale) {
  const int wid  = (blockIdx.x * blockDim.x + threadIdx.x) >> 5;
  const int lane = threadIdx.x & 31;
  const int h  = wid % NHx;
  const int bs = wid / NHx;          // b*SEQ + s
  const int s  = bs % SEQ;
  const int b  = bs / SEQ;

  const float* x = X + ((size_t)bs * NHx + h) * HD;
  float v[4];
  float ss = 0.f;
#pragma unroll
  for (int j = 0; j < 4; ++j) { v[j] = x[lane * 4 + j]; ss += v[j] * v[j]; }
#pragma unroll
  for (int msk = 16; msk >= 1; msk >>= 1) ss += __shfl_xor(ss, msk, 32);
  const float inv = rsqrtf(ss * (1.0f / (float)HD) + 1e-6f);

  const float* cp = cosb + (size_t)bs * HD;
  const float* sp = sinb + (size_t)bs * HD;
  const float sign = (lane < 16) ? -1.f : 1.f;  // rotate_half sign
  __bf16* y = Y + (((size_t)b * NHx + h) * SEQ + s) * HD;
#pragma unroll
  for (int j = 0; j < 4; ++j) {
    int d = lane * 4 + j;
    float xn = v[j] * inv * nw[d];
    float xp = __shfl_xor(xn, 16, 32);           // value at d +/- 64
    y[d] = (__bf16)((xn * cp[d] + sign * xp * sp[d]) * outscale);
  }
}

// ---------------------------------------------------------------------------
// V: fp32 (B,S,NKV,HD) -> bf16 transposed (B,NKV,HD,S)
// ---------------------------------------------------------------------------
__global__ __launch_bounds__(256) void v_trans(
    const float* __restrict__ X, __bf16* __restrict__ Y) {
  const int wid  = (blockIdx.x * blockDim.x + threadIdx.x) >> 5;
  const int lane = threadIdx.x & 31;
  const int h  = wid % NKV;
  const int bs = wid / NKV;
  const int s  = bs % SEQ;
  const int b  = bs / SEQ;
  const float* x = X + ((size_t)bs * NKV + h) * HD;
  __bf16* y = Y + (((size_t)b * NKV + h) * HD) * SEQ + s;
#pragma unroll
  for (int j = 0; j < 4; ++j) {
    int d = lane * 4 + j;
    y[(size_t)d * SEQ] = (__bf16)x[d];
  }
}

// ---------------------------------------------------------------------------
// Flash attention: one wave handles 16 queries of one (b,h); streams keys in
// blocks of 32; QK^T and PV via bf16 WMMA; online softmax in fp32.
// Output written directly as bf16 for the O-projection GEMM.
// ---------------------------------------------------------------------------
__global__ __launch_bounds__(32) void attn(
    const __bf16* __restrict__ Qb,  // [B,NH,S,HD], pre-scaled by 1/sqrt(HD)
    const __bf16* __restrict__ Kb,  // [B,NKV,S,HD]
    const __bf16* __restrict__ Vb,  // [B,NKV,HD,S]
    __bf16* __restrict__ AO) {      // [B,S,NH*HD]
  __shared__ __bf16 P[16 * 32];
  const int lane = threadIdx.x;
  const int qt = blockIdx.x % (SEQ / 16);
  const int bh = blockIdx.x / (SEQ / 16);
  const int h  = bh % NH;
  const int b  = bh / NH;
  const int kv = h / (NH / NKV);

  const __bf16* qbase = Qb + (((size_t)b * NH + h) * SEQ + qt * 16) * HD;
  const __bf16* kbase = Kb + ((size_t)b * NKV + kv) * SEQ * HD;
  const __bf16* vbase = Vb + ((size_t)b * NKV + kv) * HD * SEQ;

  // Q A-fragments, resident for whole key loop (4 chunks of K=32 over HD=128)
  v16bf qf[4];
  {
    const __bf16* qr = qbase + (size_t)(lane & 15) * HD + (lane >> 4) * 8;
#pragma unroll
    for (int c = 0; c < 4; ++c) qf[c] = frag_a_bf(qr + c * 32);
  }

  v8f o[8];
  float m[8], l[8];
#pragma unroll
  for (int t = 0; t < 8; ++t) o[t] = (v8f){};
#pragma unroll
  for (int r = 0; r < 8; ++r) { m[r] = -1e30f; l[r] = 0.f; }

  for (int k0 = 0; k0 < SEQ; k0 += 32) {
    // ---- scores S(16x32) = Q(16x128) * K_block(32x128)^T ----
    v8f s0 = (v8f){}, s1 = (v8f){};
    {
      const int n  = lane & 15;
      const int ko = (lane >> 4) * 8;
#pragma unroll
      for (int c = 0; c < 4; ++c) {
        v16bf b0 = frag_b_bf(kbase + (size_t)(k0 + n) * HD + c * 32 + ko);
        v16bf b1 = frag_b_bf(kbase + (size_t)(k0 + 16 + n) * HD + c * 32 + ko);
        s0 = wmma_bf16(qf[c], b0, s0);
        s1 = wmma_bf16(qf[c], b1, s1);
      }
    }
    // ---- online softmax (row = r + 8*(lane>=16), cols across 16-lane half) ----
    float bm[8];
#pragma unroll
    for (int r = 0; r < 8; ++r) bm[r] = fmaxf(s0[r], s1[r]);
#pragma unroll
    for (int msk = 8; msk >= 1; msk >>= 1)
#pragma unroll
      for (int r = 0; r < 8; ++r) bm[r] = fmaxf(bm[r], __shfl_xor(bm[r], msk, 32));

    float fac[8];
#pragma unroll
    for (int r = 0; r < 8; ++r) {
      float nm = fmaxf(m[r], bm[r]);
      fac[r] = __expf(m[r] - nm);
      m[r] = nm;
    }
#pragma unroll
    for (int t = 0; t < 8; ++t)
#pragma unroll
      for (int r = 0; r < 8; ++r) o[t][r] *= fac[r];

    float psum[8];
#pragma unroll
    for (int r = 0; r < 8; ++r) {
      float p0 = __expf(s0[r] - m[r]);
      float p1 = __expf(s1[r] - m[r]);
      psum[r] = p0 + p1;
      int row = r + 8 * (lane >> 4);
      P[row * 32 + (lane & 15)]      = (__bf16)p0;
      P[row * 32 + 16 + (lane & 15)] = (__bf16)p1;
    }
#pragma unroll
    for (int msk = 8; msk >= 1; msk >>= 1)
#pragma unroll
      for (int r = 0; r < 8; ++r) psum[r] += __shfl_xor(psum[r], msk, 32);
#pragma unroll
    for (int r = 0; r < 8; ++r) l[r] = l[r] * fac[r] + psum[r];

    __syncthreads();  // DS-wait: C-layout -> A-layout transpose through LDS
    v16bf pf = frag_a_bf(&P[(lane & 15) * 32 + (lane >> 4) * 8]);

    // ---- O(16x128) += P(16x32) * V_block(32x128) ----
    {
      const int n  = lane & 15;
      const int ko = (lane >> 4) * 8;
#pragma unroll
      for (int t = 0; t < 8; ++t) {
        v16bf vf = frag_b_bf(vbase + (size_t)(t * 16 + n) * SEQ + k0 + ko);
        o[t] = wmma_bf16(pf, vf, o[t]);
      }
    }
  }

  // epilogue: normalize rows, write bf16 (B,S,NH*HD) for the O-projection
#pragma unroll
  for (int r = 0; r < 8; ++r) {
    float invl = 1.0f / l[r];
    int row = qt * 16 + r + 8 * (lane >> 4);
    __bf16* ao = AO + ((size_t)b * SEQ + row) * (NH * HD) + h * HD;
#pragma unroll
    for (int t = 0; t < 8; ++t) ao[t * 16 + (lane & 15)] = (__bf16)(o[t][r] * invl);
  }
}

// ---------------------------------------------------------------------------
extern "C" void kernel_launch(void* const* d_in, const int* in_sizes, int n_in,
                              void* d_out, int out_size, void* d_ws, size_t ws_size,
                              hipStream_t stream) {
  const float* hs   = (const float*)d_in[0];
  const float* cosb = (const float*)d_in[1];
  const float* sinb = (const float*)d_in[2];
  const float* Wq   = (const float*)d_in[3];
  const float* Wk   = (const float*)d_in[4];
  const float* Wv   = (const float*)d_in[5];
  const float* Wo   = (const float*)d_in[6];
  const float* qw   = (const float*)d_in[7];
  const float* kw   = (const float*)d_in[8];
  float* out = (float*)d_out;

  char* ws = (char*)d_ws;
  __bf16* hb  = (__bf16*)(ws);                   // 16,777,216 B
  __bf16* Wqb = (__bf16*)(ws + 16777216);        //  8,388,608 B
  __bf16* Wkb = (__bf16*)(ws + 25165824);        //  4,194,304 B
  __bf16* Wvb = (__bf16*)(ws + 29360128);        //  4,194,304 B
  __bf16* Wob = (__bf16*)(ws + 33554432);        //  8,388,608 B
  float*  Qf  = (float*)(ws + 41943040);         // 33,554,432 B
  __bf16* AOb = (__bf16*)(ws + 41943040);        // aliases Qf (dead after normrope)
  float*  Kf  = (float*)(ws + 75497472);         // 16,777,216 B
  float*  Vf  = (float*)(ws + 92274688);         // 16,777,216 B
  __bf16* Qb  = (__bf16*)(ws + 109051904);       // 16,777,216 B
  __bf16* Kb  = (__bf16*)(ws + 125829120);       //  8,388,608 B
  __bf16* Vb  = (__bf16*)(ws + 134217728);       //  8,388,608 B

  const int M = BATCH * SEQ;  // 4096
  const float qscale = 0.08838834764831845f;  // 1/sqrt(128)

  // One-time fp32 -> bf16 precision change for activations and weights
  cvt_f32_bf16<<<(M * HID) / 2048, 256, 0, stream>>>(hs, hb);
  cvt_f32_bf16<<<(HID * NH * HD) / 2048, 256, 0, stream>>>(Wq, Wqb);
  cvt_f32_bf16<<<(HID * NKV * HD) / 2048, 256, 0, stream>>>(Wk, Wkb);
  cvt_f32_bf16<<<(HID * NKV * HD) / 2048, 256, 0, stream>>>(Wv, Wvb);
  cvt_f32_bf16<<<(HID * NH * HD) / 2048, 256, 0, stream>>>(Wo, Wob);

  // QKV projections (pure bf16 WMMA, fp32 accumulate)
  {
    dim3 gq((NH * HD) / 256, M / 64);
    dim3 gkv((NKV * HD) / 256, M / 64);
    gemm_nt_bf16<<<gq, 256, 0, stream>>>(hb, Wqb, Qf, M, NH * HD, HID);
    gemm_nt_bf16<<<gkv, 256, 0, stream>>>(hb, Wkb, Kf, M, NKV * HD, HID);
    gemm_nt_bf16<<<gkv, 256, 0, stream>>>(hb, Wvb, Vf, M, NKV * HD, HID);
  }

  // RMSNorm + RoPE (+ fold softmax scale into Q), emit bf16 head-major
  normrope<<<(M * NH) / 8, 256, 0, stream>>>(Qf, qw, cosb, sinb, Qb, NH, qscale);
  normrope<<<(M * NKV) / 8, 256, 0, stream>>>(Kf, kw, cosb, sinb, Kb, NKV, 1.0f);
  v_trans<<<(M * NKV) / 8, 256, 0, stream>>>(Vf, Vb);

  // Flash attention (writes bf16 attention output)
  attn<<<BATCH * NH * (SEQ / 16), 32, 0, stream>>>(Qb, Kb, Vb, AOb);

  // Output projection
  {
    dim3 go(HID / 256, M / 64);
    gemm_nt_bf16<<<go, 256, 0, stream>>>(AOb, Wob, out, M, HID, NH * HD);
  }
}
